// MllamaTextCrossAttention_9680856285243
// MI455X (gfx1250) — compile-verified, matrix operand order checked by reading, not run on hardware
//
#include <hip/hip_runtime.h>
#include <cstdint>

#define NH      32
#define NKV     8
#define HD      128
#define HDIM    4096
#define BB      4
#define SQy     1024
#define SKV     1601
#define SKV_PAD 1664
#define QSZ     (NH * HD)   /* 4096 */
#define MKV     (BB * SKV)  /* 6404 */
#define ATT_SCALE 0.08838834764831845f
#define EPSF    1e-5f
#define BKP     34          /* 32 + 2 pad (bank-conflict-free fragment reads) */
#define AOP     40          /* bf16 A-tile stride for 16B-aligned async LDS rows */
#define KSTR    136         /* attention K-tile stride: 16B-aligned rows */

typedef __attribute__((ext_vector_type(16))) __bf16 v16bf;
typedef __attribute__((ext_vector_type(8)))  float  v8f;

// Fragment loader for 16-bit WMMA A/B operands from LDS.
// Lane m/n = lane&15 is folded into the row pointer by the caller.
// Per ISA 16-bit A 16x32 layout: element pair (2r,2r+1) holds K = 2(r&3)+16(r>>2)+8*half.
static __device__ __forceinline__ v16bf ldfrag(const __bf16* p) {
  const int half8 = (threadIdx.x & 16) ? 8 : 0;
  v16bf f;
#pragma unroll
  for (int r = 0; r < 8; ++r) {
    const int k = ((r & 3) << 1) + ((r >> 2) << 4) + half8;
    f[2 * r]     = p[k];
    f[2 * r + 1] = p[k + 1];
  }
  return f;
}

static __device__ __forceinline__ v8f wmma_bf16(v16bf a, v16bf b, v8f c) {
  return __builtin_amdgcn_wmma_f32_16x16x32_bf16(false, a, false, b, (short)0, c,
                                                 false, false);
}

// Async copy of 16 bytes global->LDS (CDNA5 ASYNCcnt path). lds = low 32 bits of
// a generic pointer to __shared__ (flat LDS aperture keeps the offset in [31:0]).
static __device__ __forceinline__ void async_b128(unsigned lds, const void* g) {
  asm volatile("global_load_async_to_lds_b128 %0, %1, off"
               :: "v"(lds), "v"((unsigned long long)g) : "memory");
}
static __device__ __forceinline__ void wait_async0() {
  asm volatile("s_wait_asynccnt 0x0" ::: "memory");
}

// Stage a 128x32 f32 tile from global, convert to bf16, into LDS [128][BKP].
static __device__ __forceinline__ void stage_f32(__bf16* dst, const float* src,
                                                 long row0, int k0, int ld,
                                                 long nvalid, int t) {
#pragma unroll
  for (int i = 0; i < 4; ++i) {
    const int row = (t >> 3) + i * 32;
    const int col = (t & 7) * 4;
    const long r  = row0 + row;
    float4 v = make_float4(0.f, 0.f, 0.f, 0.f);
    if (r < nvalid) v = *(const float4*)(src + r * (long)ld + k0 + col);
    __bf16* d = dst + row * BKP + col;
    d[0] = (__bf16)v.x; d[1] = (__bf16)v.y; d[2] = (__bf16)v.z; d[3] = (__bf16)v.w;
  }
}

// ---------------------------------------------------------------------------
// Kernel 1: Q = hidden @ wqkv[:4096]^T, fused per-head RMS norm, bf16 out
// Grid: (32 n-tiles == heads, 32 m-tiles). Block 256 = 8 waves.
// ---------------------------------------------------------------------------
__global__ __launch_bounds__(256) void gemm_q_kernel(
    const float* __restrict__ A, const float* __restrict__ W,
    const float* __restrict__ qw, __bf16* __restrict__ Qb) {
  __shared__ __bf16 As[128 * BKP];
  __shared__ __bf16 Bs[128 * BKP];
  __shared__ float  rowsq[128];
  __shared__ float  rscale[128];
  const int t = threadIdx.x;
  const int wave = t >> 5, lane = t & 31, l15 = lane & 15;
  const int half8 = (t & 16) ? 8 : 0;
  const int mw = (wave >> 1) * 32, nw = (wave & 1) * 64;
  const long m0 = (long)blockIdx.y * 128;
  const int  n0 = blockIdx.x * 128;
  if (t < 128) rowsq[t] = 0.f;
  const v8f vzero = {};
  v8f acc[2][4];
#pragma unroll
  for (int i = 0; i < 2; ++i)
#pragma unroll
    for (int j = 0; j < 4; ++j) acc[i][j] = vzero;

  for (int k0 = 0; k0 < HDIM; k0 += 32) {
    __syncthreads();
    stage_f32(As, A, m0, k0, HDIM, (long)1 << 60, t);
    stage_f32(Bs, W, n0, k0, HDIM, (long)1 << 60, t);
    if (k0 + 32 < HDIM) {
      __builtin_prefetch(A + (m0 + (t >> 3)) * (long)HDIM + k0 + 32, 0, 1);
      __builtin_prefetch(W + (n0 + (t >> 3)) * (long)HDIM + k0 + 32, 0, 1);
    }
    __syncthreads();
    const v16bf a0 = ldfrag(&As[(mw + l15) * BKP]);
    const v16bf a1 = ldfrag(&As[(mw + 16 + l15) * BKP]);
#pragma unroll
    for (int j = 0; j < 4; ++j) {
      const v16bf b = ldfrag(&Bs[(nw + j * 16 + l15) * BKP]);
      acc[0][j] = wmma_bf16(a0, b, acc[0][j]);
      acc[1][j] = wmma_bf16(a1, b, acc[1][j]);
    }
  }
  // RMS-norm: tile's 128 columns == one full head.
#pragma unroll
  for (int i = 0; i < 2; ++i)
#pragma unroll
    for (int r = 0; r < 8; ++r) {
      float ss = 0.f;
#pragma unroll
      for (int j = 0; j < 4; ++j) { const float v = acc[i][j][r]; ss += v * v; }
      ss += __shfl_xor(ss, 8, 32);
      ss += __shfl_xor(ss, 4, 32);
      ss += __shfl_xor(ss, 2, 32);
      ss += __shfl_xor(ss, 1, 32);
      if (l15 == 0) atomicAdd(&rowsq[mw + i * 16 + r + half8], ss);
    }
  __syncthreads();
  if (t < 128) rscale[t] = rsqrtf(rowsq[t] * (1.f / 128.f) + EPSF);
  __syncthreads();
  const int head = blockIdx.x;
#pragma unroll
  for (int i = 0; i < 2; ++i)
#pragma unroll
    for (int j = 0; j < 4; ++j)
#pragma unroll
      for (int r = 0; r < 8; ++r) {
        const int row = mw + i * 16 + r + half8;
        const long m = m0 + row;
        const int bb = (int)(m >> 10), s = (int)(m & 1023);
        const int d = nw + j * 16 + l15;
        const float val = acc[i][j][r] * rscale[row] * qw[d];
        Qb[((long)(bb * NH + head) * SQy + s) * HD + d] = (__bf16)val;
      }
}

// ---------------------------------------------------------------------------
// Kernel 2: K,V = cross @ wqkv[4096:]^T; k-norm fused for n-tiles < 8.
// Grid: (16 n-tiles, 51 m-tiles).
// ---------------------------------------------------------------------------
__global__ __launch_bounds__(256) void gemm_kv_kernel(
    const float* __restrict__ A, const float* __restrict__ Wkv,
    const float* __restrict__ kw, __bf16* __restrict__ Kb,
    __bf16* __restrict__ Vb) {
  __shared__ __bf16 As[128 * BKP];
  __shared__ __bf16 Bs[128 * BKP];
  __shared__ float  rowsq[128];
  __shared__ float  rscale[128];
  const int t = threadIdx.x;
  const int wave = t >> 5, lane = t & 31, l15 = lane & 15;
  const int half8 = (t & 16) ? 8 : 0;
  const int mw = (wave >> 1) * 32, nw = (wave & 1) * 64;
  const long m0 = (long)blockIdx.y * 128;
  const int  n0 = blockIdx.x * 128;
  const bool isK = blockIdx.x < 8;
  if (t < 128) rowsq[t] = 0.f;
  const v8f vzero = {};
  v8f acc[2][4];
#pragma unroll
  for (int i = 0; i < 2; ++i)
#pragma unroll
    for (int j = 0; j < 4; ++j) acc[i][j] = vzero;

  for (int k0 = 0; k0 < HDIM; k0 += 32) {
    __syncthreads();
    stage_f32(As, A, m0, k0, HDIM, (long)MKV, t);
    stage_f32(Bs, Wkv, n0, k0, HDIM, (long)1 << 60, t);
    __syncthreads();
    const v16bf a0 = ldfrag(&As[(mw + l15) * BKP]);
    const v16bf a1 = ldfrag(&As[(mw + 16 + l15) * BKP]);
#pragma unroll
    for (int j = 0; j < 4; ++j) {
      const v16bf b = ldfrag(&Bs[(nw + j * 16 + l15) * BKP]);
      acc[0][j] = wmma_bf16(a0, b, acc[0][j]);
      acc[1][j] = wmma_bf16(a1, b, acc[1][j]);
    }
  }
  if (isK) {
#pragma unroll
    for (int i = 0; i < 2; ++i)
#pragma unroll
      for (int r = 0; r < 8; ++r) {
        float ss = 0.f;
#pragma unroll
        for (int j = 0; j < 4; ++j) { const float v = acc[i][j][r]; ss += v * v; }
        ss += __shfl_xor(ss, 8, 32);
        ss += __shfl_xor(ss, 4, 32);
        ss += __shfl_xor(ss, 2, 32);
        ss += __shfl_xor(ss, 1, 32);
        if (l15 == 0) atomicAdd(&rowsq[mw + i * 16 + r + half8], ss);
      }
  }
  __syncthreads();
  if (t < 128) rscale[t] = isK ? rsqrtf(rowsq[t] * (1.f / 128.f) + EPSF) : 1.f;
  __syncthreads();
  const int headk = blockIdx.x & 7;
#pragma unroll
  for (int i = 0; i < 2; ++i)
#pragma unroll
    for (int j = 0; j < 4; ++j)
#pragma unroll
      for (int r = 0; r < 8; ++r) {
        const int row = mw + i * 16 + r + half8;
        const long m = m0 + row;
        if (m >= MKV) continue;
        const int mi = (int)m;
        const int bb = mi / SKV;
        const int s  = mi - bb * SKV;
        const int d  = nw + j * 16 + l15;
        float val = acc[i][j][r] * rscale[row];
        const long idx = ((long)(bb * NKV + headk) * SKV_PAD + s) * HD + d;
        if (isK) { val *= kw[d]; Kb[idx] = (__bf16)val; }
        else     { Vb[idx] = (__bf16)val; }
      }
}

// ---------------------------------------------------------------------------
// Kernel 3: flash attention. Grid (8 q-tiles, 32 heads, 4 batch).
// Block 256 = 8 waves; each wave owns 16 Q rows x 128 d.
// K tile staged via async global->LDS copies (ASYNCcnt); V transposed on VALU.
// ---------------------------------------------------------------------------
__global__ __launch_bounds__(256) void flash_attn_kernel(
    const __bf16* __restrict__ Qb, const __bf16* __restrict__ Kb,
    const __bf16* __restrict__ Vb, __bf16* __restrict__ Ob) {
  __shared__ __bf16 Qs[128 * BKP];     //  8.7 KB
  __shared__ __bf16 Ks[64 * KSTR];     // 17.4 KB (16B-aligned rows)
  __shared__ __bf16 Vt[128 * 66];      // 16.9 KB (V transposed: [d][s])
  __shared__ __bf16 Ps[8 * 16 * 66];   // 16.9 KB (per-wave P)
  const int t = threadIdx.x;
  const int wave = t >> 5, lane = t & 31, l15 = lane & 15;
  const int half8 = (t & 16) ? 8 : 0;
  const int b = blockIdx.z, h = blockIdx.y, kvh = h >> 2;
  const int q0 = blockIdx.x * 128;
  const __bf16* Qg = Qb + ((long)(b * NH + h) * SQy + q0) * HD;
  const __bf16* Kg = Kb + (long)(b * NKV + kvh) * SKV_PAD * HD;
  const __bf16* Vg = Vb + (long)(b * NKV + kvh) * SKV_PAD * HD;
  const v8f vzero = {};
  const unsigned ksbase = (unsigned)(unsigned long long)(&Ks[0]);

  // Load this wave's Q fragments (16 rows x 128 d = 4 K-chunks), keep in regs.
  v16bf qf[4];
  for (int kc = 0; kc < 4; ++kc) {
    __syncthreads();
#pragma unroll
    for (int i = 0; i < 4; ++i) {
      const int row = (t >> 3) + i * 32;
      const int col = (t & 7) * 4;
      const __bf16* s = Qg + (long)row * HD + kc * 32 + col;
      __bf16* d = &Qs[row * BKP + col];
      d[0] = s[0]; d[1] = s[1]; d[2] = s[2]; d[3] = s[3];
    }
    __syncthreads();
    qf[kc] = ldfrag(&Qs[(wave * 16 + l15) * BKP]);
  }

  v8f oacc[8];
#pragma unroll
  for (int j = 0; j < 8; ++j) oacc[j] = vzero;
  float m_run[8], l_run[8];
#pragma unroll
  for (int r = 0; r < 8; ++r) { m_run[r] = -1e30f; l_run[r] = 0.f; }
  __bf16* Pw = &Ps[wave * 16 * 66];

  for (int st = 0; st < SKV_PAD / 64; ++st) {  // 26 tiles of 64 kv rows
    const int s0 = st * 64;
    __syncthreads();
    // Async-stage K tile [64][128] bf16: 1024 x 16B ops, 4 per thread.
#pragma unroll
    for (int i = 0; i < 4; ++i) {
      const int linear = t + i * 256;
      const int row = linear >> 4;       // 0..63
      const int blk = linear & 15;       // 16B block within row
      const unsigned lds = ksbase + (unsigned)(row * (KSTR * 2) + blk * 16);
      async_b128(lds, Kg + (long)(s0 + row) * HD + blk * 8);
    }
    {  // V tile transposed into Vt[d][s] (VALU path; scatter can't be async)
      const int row = t >> 2;
      const int cg  = t & 3;
#pragma unroll
      for (int i = 0; i < 8; ++i) {
        const int col = cg * 32 + i * 4;
        const __bf16* vs = Vg + (long)(s0 + row) * HD + col;
        Vt[(col + 0) * 66 + row] = vs[0];
        Vt[(col + 1) * 66 + row] = vs[1];
        Vt[(col + 2) * 66 + row] = vs[2];
        Vt[(col + 3) * 66 + row] = vs[3];
      }
    }
    wait_async0();
    __syncthreads();

    // S = Q * K^T  (16 x 64)
    v8f sacc[4];
#pragma unroll
    for (int j = 0; j < 4; ++j) {
      v8f c = vzero;
#pragma unroll
      for (int kc = 0; kc < 4; ++kc) {
        const v16bf kb = ldfrag(&Ks[(j * 16 + l15) * KSTR + kc * 32]);
        c = wmma_bf16(qf[kc], kb, c);
      }
      sacc[j] = c;
    }

    // Online softmax (rows in C layout: row = r + half8, col = j*16 + l15)
    float mnew[8], alpha[8];
#pragma unroll
    for (int r = 0; r < 8; ++r) {
      float mx = m_run[r];
#pragma unroll
      for (int j = 0; j < 4; ++j) {
        const float v = (float)sacc[j][r] * ATT_SCALE;
        sacc[j][r] = v;
        const int col = s0 + j * 16 + l15;
        if (col < SKV) mx = fmaxf(mx, v);
      }
      mx = fmaxf(mx, __shfl_xor(mx, 8, 32));
      mx = fmaxf(mx, __shfl_xor(mx, 4, 32));
      mx = fmaxf(mx, __shfl_xor(mx, 2, 32));
      mx = fmaxf(mx, __shfl_xor(mx, 1, 32));
      mnew[r]  = mx;
      alpha[r] = __expf(m_run[r] - mx);
      m_run[r] = mx;
    }
#pragma unroll
    for (int r = 0; r < 8; ++r) {
      float ps = 0.f;
#pragma unroll
      for (int j = 0; j < 4; ++j) {
        const int col = s0 + j * 16 + l15;
        const float p = (col < SKV) ? __expf((float)sacc[j][r] - mnew[r]) : 0.f;
        ps += p;
        Pw[(r + half8) * 66 + j * 16 + l15] = (__bf16)p;
      }
      ps += __shfl_xor(ps, 8, 32);
      ps += __shfl_xor(ps, 4, 32);
      ps += __shfl_xor(ps, 2, 32);
      ps += __shfl_xor(ps, 1, 32);
      l_run[r] = l_run[r] * alpha[r] + ps;
    }
#pragma unroll
    for (int j8 = 0; j8 < 8; ++j8)
#pragma unroll
      for (int r = 0; r < 8; ++r) oacc[j8][r] *= alpha[r];

    // Make per-wave P stores visible before fragment reads (CDNA5 DS counter)
    asm volatile("s_wait_dscnt 0x0" ::: "memory");

    // O += P * V
    const v16bf p0 = ldfrag(&Pw[l15 * 66 + 0]);
    const v16bf p1 = ldfrag(&Pw[l15 * 66 + 32]);
#pragma unroll
    for (int j8 = 0; j8 < 8; ++j8) {
      const v16bf v0 = ldfrag(&Vt[(j8 * 16 + l15) * 66 + 0]);
      const v16bf v1 = ldfrag(&Vt[(j8 * 16 + l15) * 66 + 32]);
      oacc[j8] = wmma_bf16(p0, v0, oacc[j8]);
      oacc[j8] = wmma_bf16(p1, v1, oacc[j8]);
    }
  }

  // Epilogue: normalize and store bf16 to Ob[b][s][h][d]
  float inv[8];
#pragma unroll
  for (int r = 0; r < 8; ++r) inv[r] = 1.f / l_run[r];
#pragma unroll
  for (int j8 = 0; j8 < 8; ++j8)
#pragma unroll
    for (int r = 0; r < 8; ++r) {
      const int srow = q0 + wave * 16 + r + half8;
      const int d    = j8 * 16 + l15;
      const float val = oacc[j8][r] * inv[r];
      Ob[((long)(b * SQy + srow) * NH + h) * HD + d] = (__bf16)val;
    }
}

// ---------------------------------------------------------------------------
// Kernel 4: out = attn_out @ wo^T (f32 out). Grid (32, 32).
// bf16 A tile staged via async global->LDS copies (ASYNCcnt).
// ---------------------------------------------------------------------------
__global__ __launch_bounds__(256) void gemm_o_kernel(
    const __bf16* __restrict__ A, const float* __restrict__ W,
    float* __restrict__ out) {
  __shared__ __bf16 As[128 * AOP];   // 16B-aligned rows for async b128
  __shared__ __bf16 Bs[128 * BKP];
  const int t = threadIdx.x;
  const int wave = t >> 5, lane = t & 31, l15 = lane & 15;
  const int half8 = (t & 16) ? 8 : 0;
  const int mw = (wave >> 1) * 32, nw = (wave & 1) * 64;
  const long m0 = (long)blockIdx.y * 128;
  const int  n0 = blockIdx.x * 128;
  const unsigned asbase = (unsigned)(unsigned long long)(&As[0]);
  const v8f vzero = {};
  v8f acc[2][4];
#pragma unroll
  for (int i = 0; i < 2; ++i)
#pragma unroll
    for (int j = 0; j < 4; ++j) acc[i][j] = vzero;

  for (int k0 = 0; k0 < HDIM; k0 += 32) {
    __syncthreads();
    // Async-stage A tile [128][32] bf16: 512 x 16B ops, 2 per thread.
#pragma unroll
    for (int i = 0; i < 2; ++i) {
      const int linear = t + i * 256;
      const int row = linear >> 2;     // 0..127
      const int blk = linear & 3;      // 16B block (8 bf16)
      const unsigned lds = asbase + (unsigned)(row * (AOP * 2) + blk * 16);
      async_b128(lds, A + (m0 + row) * (long)HDIM + k0 + blk * 8);
    }
    stage_f32(Bs, W, n0, k0, HDIM, (long)1 << 60, t);
    wait_async0();
    __syncthreads();
    const v16bf a0 = ldfrag(&As[(mw + l15) * AOP]);
    const v16bf a1 = ldfrag(&As[(mw + 16 + l15) * AOP]);
#pragma unroll
    for (int j = 0; j < 4; ++j) {
      const v16bf b = ldfrag(&Bs[(nw + j * 16 + l15) * BKP]);
      acc[0][j] = wmma_bf16(a0, b, acc[0][j]);
      acc[1][j] = wmma_bf16(a1, b, acc[1][j]);
    }
  }
#pragma unroll
  for (int i = 0; i < 2; ++i)
#pragma unroll
    for (int j = 0; j < 4; ++j)
#pragma unroll
      for (int r = 0; r < 8; ++r) {
        const int row = mw + i * 16 + r + half8;
        const int col = n0 + nw + j * 16 + l15;
        out[(m0 + row) * (long)HDIM + col] = acc[i][j][r];
      }
}

// ---------------------------------------------------------------------------
extern "C" void kernel_launch(void* const* d_in, const int* in_sizes, int n_in,
                              void* d_out, int out_size, void* d_ws, size_t ws_size,
                              hipStream_t stream) {
  (void)in_sizes; (void)n_in; (void)out_size; (void)ws_size;
  const float* hs   = (const float*)d_in[0];  // hidden_states        [4,1024,4096]
  const float* cs   = (const float*)d_in[1];  // cross_attn_states    [4,1601,4096]
  const float* wqkv = (const float*)d_in[2];  // [6144, 4096]
  const float* wo   = (const float*)d_in[3];  // [4096, 4096]
  const float* qw   = (const float*)d_in[4];  // [128]
  const float* kw   = (const float*)d_in[5];  // [128]
  float* out = (float*)d_out;

  const size_t qElems  = (size_t)BB * NH * SQy * HD;       // 16,777,216
  const size_t kvElems = (size_t)BB * NKV * SKV_PAD * HD;  //  6,815,744
  __bf16* Qbuf = (__bf16*)d_ws;
  __bf16* Kbuf = Qbuf + qElems;
  __bf16* Vbuf = Kbuf + kvElems;
  __bf16* Obuf = Vbuf + kvElems;

  gemm_q_kernel<<<dim3(32, 32), 256, 0, stream>>>(hs, wqkv, qw, Qbuf);
  gemm_kv_kernel<<<dim3(16, 51), 256, 0, stream>>>(cs, wqkv + (size_t)QSZ * HDIM,
                                                   kw, Kbuf, Vbuf);
  flash_attn_kernel<<<dim3(8, 32, 4), 256, 0, stream>>>(Qbuf, Kbuf, Vbuf, Obuf);
  gemm_o_kernel<<<dim3(32, 32), 256, 0, stream>>>(Obuf, wo, out);
}
